// TopKDecoder_82755429860239
// MI455X (gfx1250) — compile-verified, matrix operand order checked by reading, not run on hardware
//
#include <hip/hip_runtime.h>
#include <hip/hip_bf16.h>

#define BB 8
#define KK 8
#define VV 32000
#define HH 512
#define TT 32
#define SOS_ID 1
#define EOS_ID 2
#define MROWS (BB * KK)     // 64
#define NG (3 * HH)         // 1536
#define KT_H (HH / 32)      // 16 K-tiles of 32

typedef __attribute__((ext_vector_type(16))) _Float16 v16h;
typedef __attribute__((ext_vector_type(8)))  float    v8f;

#define NEG_INF (-__builtin_huge_valf())

// ---------------------------------------------------------------------------
// Fragment coordinate mapping for 16-bit WMMA operands (16x32 A / 32x16 B).
// Per ISA 7.12.2: lanes 0-15 / 16-31 split K ranges; VGPR p holds a K pair.
// Packed layout: frag[((tileRow*numKTiles + kTile)*32 + lane)*16 + e]
// so each lane loads 32 contiguous bytes per fragment.
// ---------------------------------------------------------------------------
__device__ __forceinline__ void frag_coords(int lane, int e, int& rowInTile, int& kInTile) {
    int p = e >> 1;
    int base = (p < 4) ? ((lane < 16) ? 0 : 8) : ((lane < 16) ? 16 : 24);
    kInTile = base + ((p & 3) << 1) + (e & 1);
    rowInTile = lane & 15;
}

// Pack B weights: src is row-major [K x Ncols] fp32 -> f16 fragments.
__global__ void pack_b(const float* __restrict__ src, _Float16* __restrict__ dst,
                       int Ncols, int numNTiles, int numKTiles) {
    long idx = blockIdx.x * (long)blockDim.x + threadIdx.x;
    long total = (long)numNTiles * numKTiles * 512;
    if (idx >= total) return;
    int e = idx & 15;
    int lane = (idx >> 4) & 31;
    long rest = idx >> 9;
    int kTile = (int)(rest % numKTiles);
    int nTile = (int)(rest / numKTiles);
    int n, kk;
    frag_coords(lane, e, n, kk);
    dst[idx] = (_Float16)src[(size_t)(kTile * 32 + kk) * Ncols + nTile * 16 + n];
}

// Pack A activations: src is row-major [M x Kcols] fp32 -> f16 fragments.
__global__ void pack_a(const float* __restrict__ src, _Float16* __restrict__ dst,
                       int numMTiles, int numKTiles, int Kcols) {
    long idx = blockIdx.x * (long)blockDim.x + threadIdx.x;
    long total = (long)numMTiles * numKTiles * 512;
    if (idx >= total) return;
    int e = idx & 15;
    int lane = (idx >> 4) & 31;
    long rest = idx >> 9;
    int kTile = (int)(rest % numKTiles);
    int mTile = (int)(rest / numKTiles);
    int m, kk;
    frag_coords(lane, e, m, kk);
    dst[idx] = (_Float16)src[(size_t)(mTile * 16 + m) * Kcols + kTile * 32 + kk];
}

// Pack x = embedding[ids] directly into A fragments (64 x 512).
__global__ void pack_x(const float* __restrict__ emb, const int* __restrict__ ids,
                       _Float16* __restrict__ dst) {
    int idx = blockIdx.x * blockDim.x + threadIdx.x;
    if (idx >= 4 * KT_H * 512) return;
    int e = idx & 15;
    int lane = (idx >> 4) & 31;
    int rest = idx >> 9;
    int kTile = rest & (KT_H - 1);
    int mTile = rest / KT_H;
    int m, kk;
    frag_coords(lane, e, m, kk);
    int row = mTile * 16 + m;
    int id = ids[row];
    dst[idx] = (_Float16)emb[(size_t)id * HH + kTile * 32 + kk];
}

// ---------------------------------------------------------------------------
// Core WMMA GEMM: C[M x ldc] = Afrag(M x 32*KT) * Bfrag(32*KT x 16*numNTiles) + bias
// One wave per 16x16 output tile; 8 waves per workgroup (wave32).
// ---------------------------------------------------------------------------
template <int KT>
__global__ void gemm_wmma(const _Float16* __restrict__ aF,
                          const _Float16* __restrict__ bF,
                          const float* __restrict__ bias,
                          float* __restrict__ C, int ldc, int numNTiles) {
    const int lane = threadIdx.x & 31;
    const int wave = threadIdx.x >> 5;
    const int nTile = blockIdx.x * (blockDim.x >> 5) + wave;
    const int mTile = blockIdx.y;
    if (nTile >= numNTiles) return;  // wave-uniform: EXEC stays all-ones for WMMA

    const v16h* aB = (const v16h*)aF + (size_t)mTile * KT * 32 + lane;
    const v16h* bB = (const v16h*)bF + (size_t)nTile * KT * 32 + lane;

    __builtin_prefetch((const void*)(bB + (KT / 2) * 32), 0, 1);

    v8f acc0 = {};
    v8f acc1 = {};
#pragma unroll
    for (int k = 0; k < KT; k += 2) {
        v16h a0 = aB[(k + 0) * 32];
        v16h b0 = bB[(k + 0) * 32];
        v16h a1 = aB[(k + 1) * 32];
        v16h b1 = bB[(k + 1) * 32];
        acc0 = __builtin_amdgcn_wmma_f32_16x16x32_f16(false, a0, false, b0,
                                                      (short)0, acc0, false, false);
        acc1 = __builtin_amdgcn_wmma_f32_16x16x32_f16(false, a1, false, b1,
                                                      (short)0, acc1, false, false);
    }

    // C/D layout: VGPR r -> lanes 0-15: (M=r, N=lane); lanes 16-31: (M=8+r, N=lane-16)
    int n = nTile * 16 + (lane & 15);
    int m0 = mTile * 16 + ((lane >> 4) << 3);
    float bv = bias ? bias[n] : 0.0f;
#pragma unroll
    for (int r = 0; r < 8; ++r)
        C[(size_t)(m0 + r) * ldc + n] = acc0[r] + acc1[r] + bv;
}

// ---------------------------------------------------------------------------
// GRU elementwise gating: h' = (1-z)*n + z*h
// ---------------------------------------------------------------------------
__global__ void gru_elem(const float* __restrict__ gx, const float* __restrict__ gh,
                         const float* __restrict__ h, float* __restrict__ hNew,
                         float* __restrict__ hHistT) {
    int idx = blockIdx.x * blockDim.x + threadIdx.x;
    if (idx >= MROWS * HH) return;
    int row = idx / HH, j = idx - row * HH;
    const float* gxr = gx + (size_t)row * NG;
    const float* ghr = gh + (size_t)row * NG;
    float xr = gxr[j], xz = gxr[HH + j], xn = gxr[2 * HH + j];
    float hr = ghr[j], hz = ghr[HH + j], hn = ghr[2 * HH + j];
    float r = 1.0f / (1.0f + __expf(-(xr + hr)));
    float z = 1.0f / (1.0f + __expf(-(xz + hz)));
    float nn = tanhf(xn + r * hn);
    float v = (1.0f - z) * nn + z * h[idx];
    hNew[idx] = v;
    hHistT[idx] = v;
}

// Per-row max + log-sum-exp (for log_softmax).
__global__ void row_stats(const float* __restrict__ X, float* __restrict__ stats, int cols) {
    int row = blockIdx.x;
    const float* x = X + (size_t)row * cols;
    __shared__ float sm[256];
    float mx = NEG_INF;
    for (int i = threadIdx.x; i < cols; i += blockDim.x) mx = fmaxf(mx, x[i]);
    sm[threadIdx.x] = mx;
    __syncthreads();
    for (int s = blockDim.x / 2; s > 0; s >>= 1) {
        if ((int)threadIdx.x < s) sm[threadIdx.x] = fmaxf(sm[threadIdx.x], sm[threadIdx.x + s]);
        __syncthreads();
    }
    mx = sm[0];
    __syncthreads();
    float sum = 0.0f;
    for (int i = threadIdx.x; i < cols; i += blockDim.x) sum += __expf(x[i] - mx);
    sm[threadIdx.x] = sum;
    __syncthreads();
    for (int s = blockDim.x / 2; s > 0; s >>= 1) {
        if ((int)threadIdx.x < s) sm[threadIdx.x] += sm[threadIdx.x + s];
        __syncthreads();
    }
    if (threadIdx.x == 0) {
        stats[row * 2] = mx;
        stats[row * 2 + 1] = __logf(sm[0]);
    }
}

__device__ __forceinline__ void tk_insert(float* tv, int* tc, float v, int c) {
    if (v <= tv[KK - 1]) return;
    int i = KK - 1;
    while (i > 0 && tv[i - 1] < v) {
        tv[i] = tv[i - 1];
        tc[i] = tc[i - 1];
        --i;
    }
    tv[i] = v;
    tc[i] = c;
}

// Per-batch top-8 over K*V candidates; updates beam state, records history.
__global__ void topk_step(const float* __restrict__ logits, const float* __restrict__ stats,
                          float* __restrict__ sc, int* __restrict__ ids,
                          int* __restrict__ predBuf, int* __restrict__ predsH,
                          int* __restrict__ symsH, float* __restrict__ scsH, int t) {
    int b = blockIdx.x;
    __shared__ float sRow[KK], sMax[KK], sLse[KK];
    if ((int)threadIdx.x < KK) {
        int row = b * KK + threadIdx.x;
        sRow[threadIdx.x] = sc[row];
        sMax[threadIdx.x] = stats[row * 2];
        sLse[threadIdx.x] = stats[row * 2 + 1];
    }
    __syncthreads();

    float tv[KK];
    int tc[KK];
    for (int i = 0; i < KK; ++i) { tv[i] = NEG_INF; tc[i] = 0; }

    const int total = KK * VV;
    for (int c = threadIdx.x; c < total; c += blockDim.x) {
        int kb = c / VV;
        int v = c - kb * VV;
        float val = sRow[kb] + (logits[(size_t)(b * KK + kb) * VV + v] - sMax[kb] - sLse[kb]);
        tk_insert(tv, tc, val, c);
    }

    __shared__ float mv[256 * KK];
    __shared__ int mc[256 * KK];
    for (int i = 0; i < KK; ++i) {
        mv[threadIdx.x * KK + i] = tv[i];
        mc[threadIdx.x * KK + i] = tc[i];
    }
    __syncthreads();

    if (threadIdx.x == 0) {
        float fv[KK];
        int fc[KK];
        for (int i = 0; i < KK; ++i) { fv[i] = NEG_INF; fc[i] = 0; }
        for (int i = 0; i < 256 * KK; ++i) tk_insert(fv, fc, mv[i], mc[i]);
        for (int k = 0; k < KK; ++k) {
            int cand = fc[k];
            int sym = cand % VV;
            int pred = cand / VV + b * KK;
            int row = b * KK + k;
            predsH[t * MROWS + row] = pred;
            symsH[t * MROWS + row] = sym;
            scsH[t * MROWS + row] = fv[k];
            ids[row] = sym;
            predBuf[row] = pred;
            sc[row] = (sym == EOS_ID) ? NEG_INF : fv[k];
        }
    }
}

__global__ void gather_h(const float* __restrict__ hNew, const int* __restrict__ predBuf,
                         float* __restrict__ h) {
    int idx = blockIdx.x * blockDim.x + threadIdx.x;
    if (idx >= MROWS * HH) return;
    int row = idx / HH, j = idx - row * HH;
    h[idx] = hNew[(size_t)predBuf[row] * HH + j];
}

__global__ void init_state(const float* __restrict__ enc, float* __restrict__ h,
                           float* __restrict__ sc, int* __restrict__ ids) {
    int idx = blockIdx.x * blockDim.x + threadIdx.x;
    if (idx < MROWS * HH) {
        int row = idx / HH, j = idx - row * HH;
        h[idx] = enc[(row / KK) * HH + j];
    }
    if (idx < MROWS) {
        sc[idx] = (idx % KK == 0) ? 0.0f : NEG_INF;
        ids[idx] = SOS_ID;
    }
}

// Sort last-step raw scores per batch (stable desc) -> s_sorted + backtrace roots.
__global__ void final_sort(const float* __restrict__ scsH, float* __restrict__ outS,
                           int* __restrict__ tp) {
    int b = threadIdx.x;
    if (b >= BB) return;
    const float* v = scsH + (size_t)(TT - 1) * MROWS + b * KK;
    float vals[KK];
    int idxs[KK];
    for (int k = 0; k < KK; ++k) { vals[k] = v[k]; idxs[k] = k; }
    for (int i = 0; i < KK; ++i) {
        int best = i;
        for (int j = i + 1; j < KK; ++j)
            if (vals[j] > vals[best]) best = j;
        float tvv = vals[i]; vals[i] = vals[best]; vals[best] = tvv;
        int tii = idxs[i]; idxs[i] = idxs[best]; idxs[best] = tii;
    }
    for (int k = 0; k < KK; ++k) {
        outS[b * KK + k] = vals[k];
        tp[b * KK + k] = idxs[k] + b * KK;
    }
}

// Backtrace: sequences (T,B,K) and which beam-row feeds decoder_outputs per (t,b).
__global__ void backtrace(const int* __restrict__ tp0, const int* __restrict__ predsH,
                          const int* __restrict__ symsH, float* __restrict__ seqOut,
                          int* __restrict__ selRow) {
    int i = threadIdx.x;
    if (i >= MROWS) return;
    int b = i / KK, k = i - b * KK;
    int tp = tp0[i];
    for (int t = TT - 1; t >= 0; --t) {
        seqOut[(size_t)t * MROWS + i] = (float)symsH[t * MROWS + tp];
        if (k == 0) selRow[t * BB + b] = tp;
        tp = predsH[t * MROWS + tp];
    }
}

// Pack the 256 selected h-history rows into A fragments for the final recompute GEMM.
__global__ void pack_hist(const float* __restrict__ hHist, const int* __restrict__ selRow,
                          _Float16* __restrict__ dst) {
    int idx = blockIdx.x * blockDim.x + threadIdx.x;
    if (idx >= 16 * KT_H * 512) return;
    int e = idx & 15;
    int lane = (idx >> 4) & 31;
    int rest = idx >> 9;
    int kTile = rest & (KT_H - 1);
    int mTile = rest / KT_H;
    int m, kk;
    frag_coords(lane, e, m, kk);
    int i = mTile * 16 + m;          // i = t*B + b
    int t = i / BB;
    int srcRow = t * MROWS + selRow[i];
    dst[idx] = (_Float16)hHist[(size_t)srcRow * HH + kTile * 32 + kk];
}

// decoder_outputs[t,b,v] = logits - max - lse for the selected rows.
__global__ void write_final(const float* __restrict__ bigOut, const float* __restrict__ bigStats,
                            float* __restrict__ dOut) {
    size_t idx = blockIdx.x * (size_t)blockDim.x + threadIdx.x;
    if (idx >= (size_t)TT * BB * VV) return;
    size_t i = idx / VV;
    dOut[idx] = bigOut[idx] - bigStats[i * 2] - bigStats[i * 2 + 1];
}

// ---------------------------------------------------------------------------
extern "C" void kernel_launch(void* const* d_in, const int* in_sizes, int n_in,
                              void* d_out, int out_size, void* d_ws, size_t ws_size,
                              hipStream_t stream) {
    (void)in_sizes; (void)n_in; (void)out_size; (void)ws_size;
    const float* enc  = (const float*)d_in[0];
    const float* emb  = (const float*)d_in[1];
    const float* Wx   = (const float*)d_in[2];
    const float* Wh   = (const float*)d_in[3];
    const float* bg   = (const float*)d_in[4];
    const float* Wout = (const float*)d_in[5];
    const float* bout = (const float*)d_in[6];
    float* out = (float*)d_out;

    char* p = (char*)d_ws;
    auto carve = [&](size_t bytes) -> void* {
        void* r = (void*)p;
        p += (bytes + 255) & ~(size_t)255;
        return r;
    };

    _Float16* WxP    = (_Float16*)carve((size_t)HH * NG * 2);
    _Float16* WhP    = (_Float16*)carve((size_t)HH * NG * 2);
    _Float16* WoutP  = (_Float16*)carve((size_t)HH * VV * 2);
    _Float16* xFrag  = (_Float16*)carve((size_t)MROWS * HH * 2);
    _Float16* hFrag  = (_Float16*)carve((size_t)MROWS * HH * 2);
    _Float16* hnFrag = (_Float16*)carve((size_t)MROWS * HH * 2);
    float* gx     = (float*)carve((size_t)MROWS * NG * 4);
    float* gh     = (float*)carve((size_t)MROWS * NG * 4);
    float* h      = (float*)carve((size_t)MROWS * HH * 4);
    float* hNew   = (float*)carve((size_t)MROWS * HH * 4);
    float* hHist  = (float*)carve((size_t)TT * MROWS * HH * 4);
    float* logits = (float*)carve((size_t)MROWS * VV * 4);
    float* stats  = (float*)carve((size_t)MROWS * 2 * 4);
    float* sc     = (float*)carve((size_t)MROWS * 4);
    int* ids      = (int*)carve((size_t)MROWS * 4);
    int* predBuf  = (int*)carve((size_t)MROWS * 4);
    int* predsH   = (int*)carve((size_t)TT * MROWS * 4);
    int* symsH    = (int*)carve((size_t)TT * MROWS * 4);
    float* scsH   = (float*)carve((size_t)TT * MROWS * 4);
    int* tp       = (int*)carve((size_t)MROWS * 4);
    int* selRow   = (int*)carve((size_t)TT * BB * 4);
    _Float16* bigFrag = (_Float16*)carve((size_t)256 * HH * 2);
    float* bigOut   = (float*)carve((size_t)256 * VV * 4);
    float* bigStats = (float*)carve((size_t)256 * 2 * 4);

    const int numNT_G = NG / 16;   // 96
    const int numNT_V = VV / 16;   // 2000

    // One-time weight packing fp32 -> f16 WMMA B fragments.
    {
        long tot = (long)numNT_G * KT_H * 512;
        pack_b<<<dim3((unsigned)((tot + 255) / 256)), 256, 0, stream>>>(Wx, WxP, NG, numNT_G, KT_H);
        pack_b<<<dim3((unsigned)((tot + 255) / 256)), 256, 0, stream>>>(Wh, WhP, NG, numNT_G, KT_H);
        long totV = (long)numNT_V * KT_H * 512;
        pack_b<<<dim3((unsigned)((totV + 255) / 256)), 256, 0, stream>>>(Wout, WoutP, VV, numNT_V, KT_H);
    }
    init_state<<<dim3((MROWS * HH + 255) / 256), 256, 0, stream>>>(enc, h, sc, ids);

    for (int t = 0; t < TT; ++t) {
        pack_x<<<dim3(128), 256, 0, stream>>>(emb, ids, xFrag);
        pack_a<<<dim3(128), 256, 0, stream>>>(h, hFrag, 4, KT_H, HH);
        gemm_wmma<KT_H><<<dim3(numNT_G / 8, 4), 256, 0, stream>>>(xFrag, WxP, bg, gx, NG, numNT_G);
        gemm_wmma<KT_H><<<dim3(numNT_G / 8, 4), 256, 0, stream>>>(hFrag, WhP, nullptr, gh, NG, numNT_G);
        gru_elem<<<dim3(128), 256, 0, stream>>>(gx, gh, h, hNew, hHist + (size_t)t * MROWS * HH);
        pack_a<<<dim3(128), 256, 0, stream>>>(hNew, hnFrag, 4, KT_H, HH);
        gemm_wmma<KT_H><<<dim3(numNT_V / 8, 4), 256, 0, stream>>>(hnFrag, WoutP, bout, logits, VV, numNT_V);
        row_stats<<<dim3(MROWS), 256, 0, stream>>>(logits, stats, VV);
        topk_step<<<dim3(BB), 256, 0, stream>>>(logits, stats, sc, ids, predBuf, predsH, symsH, scsH, t);
        gather_h<<<dim3(128), 256, 0, stream>>>(hNew, predBuf, h);
    }

    float* seqOut = out + (size_t)TT * BB * VV;
    float* sSorted = seqOut + (size_t)TT * BB * KK;
    final_sort<<<dim3(1), 32, 0, stream>>>(scsH, sSorted, tp);
    backtrace<<<dim3(1), 64, 0, stream>>>(tp, predsH, symsH, seqOut, selRow);

    pack_hist<<<dim3((16 * KT_H * 512) / 256), 256, 0, stream>>>(hHist, selRow, bigFrag);
    gemm_wmma<KT_H><<<dim3(numNT_V / 8, 16), 256, 0, stream>>>(bigFrag, WoutP, bout, bigOut, VV, numNT_V);
    row_stats<<<dim3(256), 256, 0, stream>>>(bigOut, bigStats, VV);
    write_final<<<dim3((unsigned)(((size_t)TT * BB * VV + 255) / 256)), 256, 0, stream>>>(bigOut, bigStats, out);
}
